// LLMSAN_16527034155606
// MI455X (gfx1250) — compile-verified, matrix operand order checked by reading, not run on hardware
//
#include <hip/hip_runtime.h>
#include <hip/hip_bf16.h>

// ---------------- problem constants (match reference) ----------------
static constexpr int kB     = 4096;
static constexpr int kT     = 128;
static constexpr int kDSEQ  = 128;
static constexpr int kH     = 256;
static constexpr int kDOUT  = 128;
static constexpr int kNEXP  = 4;
static constexpr int kEDGES = 838860;   // int(4096*4096*0.05)
static constexpr float kSCALE = 0.1f;

typedef __attribute__((ext_vector_type(16))) __bf16 v16bf;
typedef __attribute__((ext_vector_type(8)))  float  v8f;

__device__ __forceinline__ unsigned short f2bfu(float f) {
  unsigned u = __builtin_bit_cast(unsigned, f);
  u += 0x7fffu + ((u >> 16) & 1u);                 // round-to-nearest-even
  return (unsigned short)(u >> 16);
}
__device__ __forceinline__ unsigned pk2(float a, float b) {
  return (unsigned)f2bfu(a) | ((unsigned)f2bfu(b) << 16);
}

struct BF16x16 { uint4 lo, hi; };                  // 32B == v16bf

// =====================================================================
// All-bf16 WMMA GEMM: C[M,N] = act( A[M,K](lda) @ W[N,K](ldw)^T + bias )
// A/W are pre-converted bf16 (row-major, element stride lda/ldw).
// Each wave computes a 16x64 strip: ONE A-fragment reused by FOUR
// B-fragments / accumulators (40 B per wmma from L2 instead of 64).
// All 10 b128 loads of a K-block are issued as DISTINCT live values
// before the 4 wmma ops -> register allocator gives them separate
// blocks and the scheduler retires wmma against partial s_wait_loadcnt
// (rolling software pipeline), instead of a full wait per wmma.
// Requires (N/16) % 4 == 0 (true here: N = 128/256/768/4096).
// Lane layout per ISA 7.12.2 (16-bit 16x32 operands):
//   lane l: row = l&15, half = l>>4
//   elems 0..7  <- K = k0 +    8*half .. +7   (one b128 load)
//   elems 8..15 <- K = k0 + 16+8*half .. +7   (one b128 load)
// act: 0 = none, 1 = relu, 2 = tanh
// =====================================================================
__global__ __launch_bounds__(256) void gemm_wmma_bf16(
    const unsigned short* __restrict__ A, int lda,
    const unsigned short* __restrict__ W, int ldw,
    const float* __restrict__ bias,
    float* __restrict__ C, int ldc,
    int M, int N, int K, int act)
{
  const int stripsN = N >> 6;                      // 4 N-tiles per strip
  const int strips  = (M >> 4) * stripsN;
  const int wave    = blockIdx.x * (blockDim.x >> 5) + (threadIdx.x >> 5);
  if (wave >= strips) return;                      // wave-uniform: EXEC stays all-1s
  const int tm   = wave / stripsN;
  const int tn0  = (wave - tm * stripsN) * 4;      // first of 4 N-tiles
  const int lane = threadIdx.x & 31;
  const int half = lane >> 4;
  const int l16  = lane & 15;

  const unsigned short* ap = A + (size_t)(tm * 16 + l16) * lda + 8 * half;
  const unsigned short* bp0 = W + (size_t)((tn0 + 0) * 16 + l16) * ldw + 8 * half;
  const unsigned short* bp1 = W + (size_t)((tn0 + 1) * 16 + l16) * ldw + 8 * half;
  const unsigned short* bp2 = W + (size_t)((tn0 + 2) * 16 + l16) * ldw + 8 * half;
  const unsigned short* bp3 = W + (size_t)((tn0 + 3) * 16 + l16) * ldw + 8 * half;

  v8f acc0 = {}, acc1 = {}, acc2 = {}, acc3 = {};
  for (int k0 = 0; k0 < K; k0 += 32) {
    // ---- issue all loads of this K-block as distinct live values ----
    BF16x16 a, b0, b1, b2, b3;
    a.lo  = *(const uint4*)(ap + k0);
    a.hi  = *(const uint4*)(ap + k0 + 16);
    b0.lo = *(const uint4*)(bp0 + k0);
    b0.hi = *(const uint4*)(bp0 + k0 + 16);
    b1.lo = *(const uint4*)(bp1 + k0);
    b1.hi = *(const uint4*)(bp1 + k0 + 16);
    b2.lo = *(const uint4*)(bp2 + k0);
    b2.hi = *(const uint4*)(bp2 + k0 + 16);
    b3.lo = *(const uint4*)(bp3 + k0);
    b3.hi = *(const uint4*)(bp3 + k0 + 16);
    __builtin_prefetch(ap + k0 + 32, 0, 1);        // global_prefetch_b8 next K-block
    // ---- 4 independent wmma (shared A) ----
    const v16bf av = __builtin_bit_cast(v16bf, a);
    acc0 = __builtin_amdgcn_wmma_f32_16x16x32_bf16(
        false, av, false, __builtin_bit_cast(v16bf, b0), (short)0, acc0, false, false);
    acc1 = __builtin_amdgcn_wmma_f32_16x16x32_bf16(
        false, av, false, __builtin_bit_cast(v16bf, b1), (short)0, acc1, false, false);
    acc2 = __builtin_amdgcn_wmma_f32_16x16x32_bf16(
        false, av, false, __builtin_bit_cast(v16bf, b2), (short)0, acc2, false, false);
    acc3 = __builtin_amdgcn_wmma_f32_16x16x32_bf16(
        false, av, false, __builtin_bit_cast(v16bf, b3), (short)0, acc3, false, false);
  }

  // C/D layout: VGPR r -> M = r + 8*half ; N = lane&15
  v8f accs[4] = {acc0, acc1, acc2, acc3};
#pragma unroll
  for (int j = 0; j < 4; ++j) {
    const int cn = (tn0 + j) * 16 + l16;
    const float bb = bias ? bias[cn] : 0.0f;
#pragma unroll
    for (int r = 0; r < 8; ++r) {
      const int cm = tm * 16 + 8 * half + r;
      float v = accs[j][r] + bb;
      if (act == 1)      v = fmaxf(v, 0.0f);
      else if (act == 2) v = tanhf(v);
      C[(size_t)cm * ldc + cn] = v;
    }
  }
}

// ---------------- f32 -> bf16 conversion (outside GEMM hot loop) -----
__global__ void cvt_bf16(const float* __restrict__ s, unsigned short* __restrict__ d,
                         long long n) {                       // n multiple of 4
  long long i = (blockIdx.x * (long long)blockDim.x + threadIdx.x) * 4;
  if (i >= n) return;
  float4 v = *(const float4*)(s + i);
  uint2 o; o.x = pk2(v.x, v.y); o.y = pk2(v.z, v.w);
  *(uint2*)(d + i) = o;
}
__global__ void cvt_bf16_rows(const float* __restrict__ src, long long srcStride,
                              unsigned short* __restrict__ dst, int rows, int cols) {
  long long i = (blockIdx.x * (long long)blockDim.x + threadIdx.x) * 4;
  if (i >= (long long)rows * cols) return;
  int r = (int)(i / cols), c = (int)(i - (long long)r * cols);
  float4 v = *(const float4*)(src + (size_t)r * srcStride + c);
  uint2 o; o.x = pk2(v.x, v.y); o.y = pk2(v.z, v.w);
  *(uint2*)(dst + i) = o;
}

// ---------------- elementwise / small kernels ----------------
__global__ void fill_f32(float* __restrict__ p, float v, long long n) {
  long long i = blockIdx.x * (long long)blockDim.x + threadIdx.x;
  if (i < n) p[i] = v;
}
__global__ void fill_u32(unsigned* __restrict__ p, unsigned v, long long n) {
  long long i = blockIdx.x * (long long)blockDim.x + threadIdx.x;
  if (i < n) p[i] = v;
}

// GRU gate: updates f32 master h AND its bf16 mirror in one pass
__global__ void gru_gate(const float* __restrict__ gi, const float* __restrict__ gh,
                         float* __restrict__ h, unsigned short* __restrict__ hb,
                         int nb, int nh) {
  int i = blockIdx.x * blockDim.x + threadIdx.x;
  if (i >= nb * nh) return;
  int b = i / nh, c = i - b * nh;
  const float* gib = gi + (size_t)b * 3 * nh;
  const float* ghb = gh + (size_t)b * 3 * nh;
  float r = 1.0f / (1.0f + expf(-(gib[c] + ghb[c])));
  float z = 1.0f / (1.0f + expf(-(gib[nh + c] + ghb[nh + c])));
  float n = tanhf(gib[2 * nh + c] + r * ghb[2 * nh + c]);
  float hv = (1.0f - z) * n + z * h[i];
  h[i]  = hv;
  hb[i] = f2bfu(hv);
}

// concat two f32 matrices into one bf16 matrix
__global__ void concat2_bf16(const float* __restrict__ a, int wa,
                             const float* __restrict__ b, int wb,
                             unsigned short* __restrict__ out, int nb) {
  int w = wa + wb;
  long long i = blockIdx.x * (long long)blockDim.x + threadIdx.x;
  if (i >= (long long)nb * w) return;
  int r = (int)(i / w), c = (int)(i - (long long)r * w);
  float v = (c < wa) ? a[(size_t)r * wa + c] : b[(size_t)r * wb + (c - wa)];
  out[i] = f2bfu(v);
}

__global__ void sub_vec_bf16(const float* __restrict__ a, const float* __restrict__ b,
                             unsigned short* __restrict__ o, long long n) {
  long long i = blockIdx.x * (long long)blockDim.x + threadIdx.x;
  if (i < n) o[i] = f2bfu(a[i] - b[i]);
}

// ---------------- top-k via histogram threshold ----------------
__global__ void sim_hist(const float* __restrict__ sim, int n, unsigned* __restrict__ hist) {
  long long i = blockIdx.x * (long long)blockDim.x + threadIdx.x;
  if (i >= (long long)n * n) return;
  int r = (int)(i / n), c = (int)(i - (long long)r * n);
  if (r == c) return;
  float v = sim[i];
  int bin = (int)floorf((v + 128.0f) * 4.0f);      // 1024 bins over [-128,128)
  bin = min(max(bin, 0), 1023);
  atomicAdd(&hist[bin], 1u);
}

__global__ void pick_threshold(const unsigned* __restrict__ hist,
                               float* __restrict__ thr, int need) {
  if (threadIdx.x == 0 && blockIdx.x == 0) {
    unsigned long long cum = 0; int bin = 0;
    for (int b = 1023; b >= 0; --b) {
      cum += hist[b];
      if (cum >= (unsigned long long)need) { bin = b; break; }
    }
    *thr = bin * 0.25f - 128.0f;                    // lower edge of chosen bin
  }
}

__global__ void compact_edges(const float* __restrict__ sim, int n,
                              const float* __restrict__ thr,
                              int* __restrict__ row, int* __restrict__ col,
                              unsigned* __restrict__ counter, int maxE) {
  long long i = blockIdx.x * (long long)blockDim.x + threadIdx.x;
  if (i >= (long long)n * n) return;
  int r = (int)(i / n), c = (int)(i - (long long)r * n);
  if (r == c) return;
  if (sim[i] >= *thr) {
    unsigned p = atomicAdd(counter, 1u);
    if (p < (unsigned)maxE) { row[p] = r; col[p] = c; }
  }
}

// ---------------- GCN ----------------
__global__ void edge_degree(const int* __restrict__ row, const int* __restrict__ col,
                            float* __restrict__ deg, int E) {
  int e = blockIdx.x * blockDim.x + threadIdx.x;
  if (e >= E) return;
  atomicAdd(&deg[row[e]], 1.0f);
  atomicAdd(&deg[col[e]], 1.0f);
}
__global__ void make_dinv(const float* __restrict__ deg, float* __restrict__ dinv, int n) {
  int i = blockIdx.x * blockDim.x + threadIdx.x;
  if (i < n) dinv[i] = rsqrtf(deg[i]);
}
__global__ void gcn_self(const float* __restrict__ xw, const float* __restrict__ dinv,
                         float* __restrict__ acc, int nb, int nh) {
  long long i = blockIdx.x * (long long)blockDim.x + threadIdx.x;
  if (i >= (long long)nb * nh) return;
  int b = (int)(i / nh);
  float d = dinv[b];
  acc[i] = xw[i] * d * d;                           // self-loop contribution
}
__global__ void gcn_scatter(const float* __restrict__ xw,
                            const int* __restrict__ row, const int* __restrict__ col,
                            const float* __restrict__ dinv,
                            float* __restrict__ acc, int E, int nh) {
  long long i = blockIdx.x * (long long)blockDim.x + threadIdx.x;
  if (i >= (long long)E * nh) return;
  int e = (int)(i / nh), f = (int)(i - (long long)e * nh);
  int r = row[e], c = col[e];
  float nrm = dinv[r] * dinv[c];
  atomicAdd(&acc[(size_t)c * nh + f], xw[(size_t)r * nh + f] * nrm); // src=r,dst=c
  atomicAdd(&acc[(size_t)r * nh + f], xw[(size_t)c * nh + f] * nrm); // src=c,dst=r
}
// relu(acc + b) emitted directly as bf16 (feeds only the next GEMM)
__global__ void gcn_epilogue_bf16(const float* __restrict__ acc, const float* __restrict__ b,
                                  unsigned short* __restrict__ out, int nb, int nh) {
  long long i = blockIdx.x * (long long)blockDim.x + threadIdx.x;
  if (i >= (long long)nb * nh) return;
  out[i] = f2bfu(fmaxf(acc[i] + b[(int)(i % nh)], 0.0f));
}

// ---------------- final expert-attention fusion ----------------
__global__ __launch_bounds__(128) void final_fuse(
    const float* __restrict__ g, const float* __restrict__ p,
    const float* __restrict__ hatt,
    const float* __restrict__ a2W, const float* __restrict__ a2b,
    const float* __restrict__ eout, float* __restrict__ pred, int nb)
{
  int b = blockIdx.x, o = threadIdx.x;
  __shared__ float wsm[kNEXP];
  if (o < kNEXP) {
    float acc = a2b[o];
    const float* hb = hatt + (size_t)b * kDOUT;
    for (int k = 0; k < kDOUT; ++k) acc += a2W[o * kDOUT + k] * hb[k];
    wsm[o] = acc;
  }
  __syncthreads();
  if (o == 0) {
    float mx = fmaxf(fmaxf(wsm[0], wsm[1]), fmaxf(wsm[2], wsm[3]));
    float s = 0.0f;
    for (int e = 0; e < kNEXP; ++e) { wsm[e] = expf(wsm[e] - mx); s += wsm[e]; }
    float inv = 1.0f / s;
    for (int e = 0; e < kNEXP; ++e) wsm[e] *= inv;
  }
  __syncthreads();
  float gg = g[(size_t)b * kDOUT + o];
  float pp = p[(size_t)b * kDOUT + o];
  float delta = 0.0f;
  for (int e = 0; e < kNEXP; ++e)
    delta += wsm[e] * eout[((size_t)e * nb + b) * kDOUT + o];
  pred[(size_t)b * kDOUT + o] = 0.5f * gg + 0.5f * pp + kSCALE * delta;
}

// =====================================================================
// host side
// =====================================================================
typedef unsigned short bf16_t;

static inline void gemm(const bf16_t* A, int lda, const bf16_t* W, int ldw,
                        const float* bias, float* C, int ldc,
                        int M, int N, int K, int act, hipStream_t s) {
  int strips = (M / 16) * (N / 64);                 // 16x64 strip per wave
  int blocks = (strips + 7) / 8;                    // 8 waves per block
  gemm_wmma_bf16<<<blocks, 256, 0, s>>>(A, lda, W, ldw, bias, C, ldc, M, N, K, act);
}
static inline int eb(long long n) { return (int)((n + 255) / 256); }

extern "C" void kernel_launch(void* const* d_in, const int* in_sizes, int n_in,
                              void* d_out, int out_size, void* d_ws, size_t ws_size,
                              hipStream_t stream) {
  (void)in_sizes; (void)n_in; (void)out_size; (void)ws_size;

  const float* input_seq = (const float*)d_in[0];
  const float* mean_emb  = (const float*)d_in[1];
  const float* prior     = (const float*)d_in[2];
  const float* Wih0 = (const float*)d_in[3];
  const float* Whh0 = (const float*)d_in[4];
  const float* bih0 = (const float*)d_in[5];
  const float* bhh0 = (const float*)d_in[6];
  const float* Wih1 = (const float*)d_in[7];
  const float* Whh1 = (const float*)d_in[8];
  const float* bih1 = (const float*)d_in[9];
  const float* bhh1 = (const float*)d_in[10];
  const float* fc_mean_W = (const float*)d_in[11];
  const float* fc_mean_b = (const float*)d_in[12];
  const float* gru_fc_W  = (const float*)d_in[13];
  const float* gru_fc_b  = (const float*)d_in[14];
  const float* gcn1_W = (const float*)d_in[15];
  const float* gcn1_b = (const float*)d_in[16];
  const float* gcn2_W = (const float*)d_in[17];
  const float* gcn2_b = (const float*)d_in[18];
  const float* prior_fc_W = (const float*)d_in[19];
  const float* prior_fc_b = (const float*)d_in[20];
  const float* expert_W = (const float*)d_in[21];
  const float* attn1_W  = (const float*)d_in[22];
  const float* attn1_b  = (const float*)d_in[23];
  const float* attn2_W  = (const float*)d_in[24];
  const float* attn2_b  = (const float*)d_in[25];

  float* pred  = (float*)d_out;
  float* g_out = (float*)d_out + (size_t)kB * kDOUT;       // gru_out_proj
  float* p_out = (float*)d_out + 2 * (size_t)kB * kDOUT;   // prior_out

  // ---- scratch carve-out (32B-aligned slots) ----
  float* base = (float*)d_ws;
  size_t off = 0;
  auto alloc = [&](size_t nf) { float* p = base + off; off += (nf + 7) & ~(size_t)7; return p; };
  auto allocb = [&](size_t ne) { return (bf16_t*)alloc((ne + 1) / 2); };  // ne bf16 elems

  // f32 buffers
  float* gi0 = alloc((size_t)kB * 3 * kH);
  float* gh0 = alloc((size_t)kB * 3 * kH);
  float* gi1 = alloc((size_t)kB * 3 * kH);
  float* gh1 = alloc((size_t)kB * 3 * kH);
  float* h1  = alloc((size_t)kB * kH);
  float* h2  = alloc((size_t)kB * kH);
  float* mean_feat = alloc((size_t)kB * kH);
  float* sim = alloc((size_t)kB * kB);
  float* xw  = alloc((size_t)kB * kH);
  float* accb = alloc((size_t)kB * kH);
  float* eout = alloc((size_t)kNEXP * kB * kDOUT);
  float* hatt = alloc((size_t)kB * kDOUT);
  // bf16 mirrors / operands
  bf16_t* h1b  = allocb((size_t)kB * kH);
  bf16_t* h2b  = allocb((size_t)kB * kH);
  bf16_t* xtb  = allocb((size_t)kB * kDSEQ);
  bf16_t* gfb  = allocb((size_t)kB * 2 * kH);
  bf16_t* priorb = allocb((size_t)kB * kH);
  bf16_t* meanb  = allocb((size_t)kB * kH);
  bf16_t* phb  = allocb((size_t)kB * kH);
  bf16_t* ph2b = allocb((size_t)kB * kH);
  bf16_t* diffb  = allocb((size_t)kB * kDOUT);
  bf16_t* fusedb = allocb((size_t)kB * 2 * kDOUT);
  // bf16 weights
  bf16_t* Wih0b = allocb((size_t)3 * kH * kDSEQ);
  bf16_t* Whh0b = allocb((size_t)3 * kH * kH);
  bf16_t* Wih1b = allocb((size_t)3 * kH * kH);
  bf16_t* Whh1b = allocb((size_t)3 * kH * kH);
  bf16_t* fcmWb = allocb((size_t)kH * kH);
  bf16_t* gfcWb = allocb((size_t)kDOUT * 2 * kH);
  bf16_t* gcn1Wb = allocb((size_t)kH * kH);
  bf16_t* gcn2Wb = allocb((size_t)kH * kH);
  bf16_t* pfcWb  = allocb((size_t)kDOUT * kH);
  bf16_t* expWb  = allocb((size_t)kNEXP * kDOUT * kDOUT);
  bf16_t* at1Wb  = allocb((size_t)kDOUT * 2 * kDOUT);
  // graph
  int* erow = (int*)alloc(kEDGES);
  int* ecol = (int*)alloc(kEDGES);
  float* deg  = alloc(kB);
  float* dinv = alloc(kB);
  unsigned* hist = (unsigned*)alloc(1024);
  unsigned* counter = (unsigned*)alloc(8);
  float* thr = alloc(8);

  const long long nBH = (long long)kB * kH;

  // ---- one-time per-call conversions (weights + static activations) ----
  auto cvt = [&](const float* s, bf16_t* d, long long n) {
    cvt_bf16<<<eb((n + 3) / 4), 256, 0, stream>>>(s, d, n);
  };
  cvt(Wih0, Wih0b, (long long)3 * kH * kDSEQ);
  cvt(Whh0, Whh0b, (long long)3 * kH * kH);
  cvt(Wih1, Wih1b, (long long)3 * kH * kH);
  cvt(Whh1, Whh1b, (long long)3 * kH * kH);
  cvt(fc_mean_W, fcmWb, (long long)kH * kH);
  cvt(gru_fc_W, gfcWb, (long long)kDOUT * 2 * kH);
  cvt(gcn1_W, gcn1Wb, (long long)kH * kH);
  cvt(gcn2_W, gcn2Wb, (long long)kH * kH);
  cvt(prior_fc_W, pfcWb, (long long)kDOUT * kH);
  cvt(expert_W, expWb, (long long)kNEXP * kDOUT * kDOUT);
  cvt(attn1_W, at1Wb, (long long)kDOUT * 2 * kDOUT);
  cvt(prior, priorb, nBH);
  cvt(mean_emb, meanb, nBH);

  // ================= GRU (layers interleaved in time) =================
  fill_f32<<<eb(nBH), 256, 0, stream>>>(h1, 0.0f, nBH);
  fill_f32<<<eb(nBH), 256, 0, stream>>>(h2, 0.0f, nBH);
  fill_u32<<<eb(nBH / 2), 256, 0, stream>>>((unsigned*)h1b, 0u, nBH / 2);
  fill_u32<<<eb(nBH / 2), 256, 0, stream>>>((unsigned*)h2b, 0u, nBH / 2);
  for (int t = 0; t < kT; ++t) {
    // layer 0
    cvt_bf16_rows<<<eb((long long)kB * kDSEQ / 4), 256, 0, stream>>>(
        input_seq + (size_t)t * kDSEQ, (long long)kT * kDSEQ, xtb, kB, kDSEQ);
    gemm(h1b, kH, Whh0b, kH, bhh0, gh0, 3 * kH, kB, 3 * kH, kH, 0, stream);
    gemm(xtb, kDSEQ, Wih0b, kDSEQ, bih0, gi0, 3 * kH, kB, 3 * kH, kDSEQ, 0, stream);
    gru_gate<<<eb(nBH), 256, 0, stream>>>(gi0, gh0, h1, h1b, kB, kH);
    // layer 1
    gemm(h2b, kH, Whh1b, kH, bhh1, gh1, 3 * kH, kB, 3 * kH, kH, 0, stream);
    gemm(h1b, kH, Wih1b, kH, bih1, gi1, 3 * kH, kB, 3 * kH, kH, 0, stream);
    gru_gate<<<eb(nBH), 256, 0, stream>>>(gi1, gh1, h2, h2b, kB, kH);
  }

  // ================= GRU head =================
  gemm(meanb, kH, fcmWb, kH, fc_mean_b, mean_feat, kH, kB, kH, kH, 1, stream);
  concat2_bf16<<<eb((long long)kB * 2 * kH), 256, 0, stream>>>(h2, kH, mean_feat, kH, gfb, kB);
  gemm(gfb, 2 * kH, gfcWb, 2 * kH, gru_fc_b, g_out, kDOUT, kB, kDOUT, 2 * kH, 0, stream);

  // ================= graph construction =================
  gemm(priorb, kH, priorb, kH, nullptr, sim, kB, kB, kB, kH, 0, stream);
  fill_u32<<<eb(1024), 256, 0, stream>>>(hist, 0u, 1024);
  fill_u32<<<1, 32, 0, stream>>>(counter, 0u, 1);
  fill_u32<<<eb(kEDGES), 256, 0, stream>>>((unsigned*)erow, 0u, kEDGES);
  fill_u32<<<eb(kEDGES), 256, 0, stream>>>((unsigned*)ecol, 0u, kEDGES);
  const long long nSim = (long long)kB * kB;
  sim_hist<<<eb(nSim), 256, 0, stream>>>(sim, kB, hist);
  pick_threshold<<<1, 32, 0, stream>>>(hist, thr, kEDGES);
  compact_edges<<<eb(nSim), 256, 0, stream>>>(sim, kB, thr, erow, ecol, counter, kEDGES);

  fill_f32<<<eb(kB), 256, 0, stream>>>(deg, 1.0f, kB);    // self loops
  edge_degree<<<eb(kEDGES), 256, 0, stream>>>(erow, ecol, deg, kEDGES);
  make_dinv<<<eb(kB), 256, 0, stream>>>(deg, dinv, kB);

  // ================= GCN layer 1 =================
  const long long nScat = (long long)kEDGES * kH;
  gemm(priorb, kH, gcn1Wb, kH, nullptr, xw, kH, kB, kH, kH, 0, stream);
  gcn_self<<<eb(nBH), 256, 0, stream>>>(xw, dinv, accb, kB, kH);
  gcn_scatter<<<eb(nScat), 256, 0, stream>>>(xw, erow, ecol, dinv, accb, kEDGES, kH);
  gcn_epilogue_bf16<<<eb(nBH), 256, 0, stream>>>(accb, gcn1_b, phb, kB, kH);

  // ================= GCN layer 2 =================
  gemm(phb, kH, gcn2Wb, kH, nullptr, xw, kH, kB, kH, kH, 0, stream);
  gcn_self<<<eb(nBH), 256, 0, stream>>>(xw, dinv, accb, kB, kH);
  gcn_scatter<<<eb(nScat), 256, 0, stream>>>(xw, erow, ecol, dinv, accb, kEDGES, kH);
  gcn_epilogue_bf16<<<eb(nBH), 256, 0, stream>>>(accb, gcn2_b, ph2b, kB, kH);

  gemm(ph2b, kH, pfcWb, kH, prior_fc_b, p_out, kDOUT, kB, kDOUT, kH, 0, stream);

  // ================= expert mixture =================
  sub_vec_bf16<<<eb((long long)kB * kDOUT), 256, 0, stream>>>(g_out, p_out, diffb,
                                                              (long long)kB * kDOUT);
  for (int e = 0; e < kNEXP; ++e)
    gemm(diffb, kDOUT, expWb + (size_t)e * kDOUT * kDOUT, kDOUT, nullptr,
         eout + (size_t)e * kB * kDOUT, kDOUT, kB, kDOUT, kDOUT, 0, stream);

  concat2_bf16<<<eb((long long)kB * 2 * kDOUT), 256, 0, stream>>>(g_out, kDOUT, p_out, kDOUT,
                                                                  fusedb, kB);
  gemm(fusedb, 2 * kDOUT, at1Wb, 2 * kDOUT, attn1_b, hatt, kDOUT,
       kB, kDOUT, 2 * kDOUT, 2, stream);

  final_fuse<<<kB, kDOUT, 0, stream>>>(g_out, p_out, hatt, attn2_W, attn2_b,
                                       eout, pred, kB);
}
